// GCN_7121055777195
// MI455X (gfx1250) — matrix operand order repaired and checked by reading, not compiled, not run on hardware
//
#include <hip/hip_runtime.h>

// ---------------------------------------------------------------------------
// GCN (2x GCNConv + linear head) for MI455X / gfx1250, wave32.
// GEMMs run on v_wmma_f32_16x16x32_bf16 with 3xBF16 error compensation
// (hi*hi + hi*lo + lo*hi) for ~fp32 accuracy at matrix-pipe speed.
// Edge aggregation: wave-per-edge float4 gather + global_atomic_add_f32.
// ---------------------------------------------------------------------------

typedef __attribute__((ext_vector_type(16))) __bf16 v16bf;
typedef __attribute__((ext_vector_type(8)))  float  v8f;

#define KDIM 128   // IN_DIM == HID_DIM == GEMM K for every layer

static inline int cdiv_i(int a, int b) { return (a + b - 1) / b; }

// ---- split fp32 weight [K x F] into transposed bf16 hi/lo [Fpad x K] -------
__global__ void gcn_wsplit_kernel(const float* __restrict__ W,
                                  __bf16* __restrict__ hiT,
                                  __bf16* __restrict__ loT,
                                  int F, int Fpad) {
    int idx = blockIdx.x * blockDim.x + threadIdx.x;
    if (idx >= Fpad * KDIM) return;
    int n = idx / KDIM;      // output column
    int k = idx % KDIM;      // K index
    float v = (n < F) ? W[k * F + n] : 0.0f;
    __bf16 hi = (__bf16)v;
    float  lo = v - (float)hi;
    hiT[(size_t)n * KDIM + k] = hi;
    loT[(size_t)n * KDIM + k] = (__bf16)lo;
}

// ---- degree with self loop --------------------------------------------------
__global__ void gcn_deg_init_kernel(float* __restrict__ deg, int n) {
    int i = blockIdx.x * blockDim.x + threadIdx.x;
    if (i < n) deg[i] = 1.0f;            // +1 self loop
}

__global__ void gcn_deg_accum_kernel(const int* __restrict__ dst,
                                     float* __restrict__ deg, int E) {
    int e = blockIdx.x * blockDim.x + threadIdx.x;
    if (e < E) atomicAdd(&deg[dst[e]], 1.0f);
}

__global__ void gcn_dinv_kernel(const float* __restrict__ deg,
                                float* __restrict__ dinv, int n) {
    int i = blockIdx.x * blockDim.x + threadIdx.x;
    if (i < n) dinv[i] = rsqrtf(deg[i]);
}

// ---- Y[nrows x Fout] = X[nrows x 128] @ W[128 x Fout] (+bias), WMMA bf16x3 --
__global__ __launch_bounds__(128)
void gcn_gemm_wmma_kernel(const float*  __restrict__ X,
                          const __bf16* __restrict__ WhiT,   // [Fpad x 128]
                          const __bf16* __restrict__ WloT,   // [Fpad x 128]
                          float* __restrict__ Y,
                          const float* __restrict__ bias,    // may be null
                          int nrows, int Fout, int Ftiles) {
    const int lane   = threadIdx.x & 31;
    const int wave   = threadIdx.x >> 5;
    const int tile_m = blockIdx.x * 4 + wave;
    if (tile_m * 16 >= nrows) return;

    const int  sub   = lane & 15;    // A: row-in-tile, B/C: column-in-tile
    const int  khalf = lane >> 4;    // 0: K 0-7 and 16-23 of chunk, 1: K 8-15 and 24-31
    const int  row   = tile_m * 16 + sub;
    const bool rok   = (row < nrows);
    const float* xrow = X + (size_t)row * KDIM;

    // ---- build A fragments (hi + lo) for all 4 K-chunks, kept in VGPRs ----
    v16bf ahi[4], alo[4];
#pragma unroll
    for (int kt = 0; kt < 4; ++kt) {
#pragma unroll
        for (int v = 0; v < 8; ++v) {
            // ISA 16-bit A 16x32 layout: VGPR v holds K pair
            //   v<4 -> 2v ; v>=4 -> 2v+8 ; lanes 16..31 add +8
            int k = kt * 32 + (v < 4 ? 2 * v : 2 * v + 8) + 8 * khalf;
            float x0 = 0.0f, x1 = 0.0f;
            if (rok) {
                float2 xv = *(const float2*)(xrow + k);
                x0 = xv.x; x1 = xv.y;
            }
            __bf16 h0 = (__bf16)x0, h1 = (__bf16)x1;
            ahi[kt][2 * v]     = h0;
            ahi[kt][2 * v + 1] = h1;
            alo[kt][2 * v]     = (__bf16)(x0 - (float)h0);
            alo[kt][2 * v + 1] = (__bf16)(x1 - (float)h1);
        }
    }

    // ---- sweep output column tiles ----
    for (int nt = 0; nt < Ftiles; ++nt) {
        const int ncol = nt * 16 + sub;
        const __bf16* whcol = WhiT + (size_t)ncol * KDIM;
        const __bf16* wlcol = WloT + (size_t)ncol * KDIM;
        v8f c = {};
#pragma unroll
        for (int kt = 0; kt < 4; ++kt) {
            v16bf bhi, blo;
#pragma unroll
            for (int v = 0; v < 8; ++v) {
                // 16-bit B 32x16 layout matching A: v<4 -> K 2v ; v>=4 -> 2v+8 ; lanes 16..31 add +8
                int k = kt * 32 + (v < 4 ? 2 * v : 2 * v + 8) + 8 * khalf;
                bhi[2 * v]     = whcol[k];
                bhi[2 * v + 1] = whcol[k + 1];
                blo[2 * v]     = wlcol[k];
                blo[2 * v + 1] = wlcol[k + 1];
            }
            c = __builtin_amdgcn_wmma_f32_16x16x32_bf16(false, ahi[kt], false, bhi,
                                                        (short)0, c, false, false);
            c = __builtin_amdgcn_wmma_f32_16x16x32_bf16(false, alo[kt], false, bhi,
                                                        (short)0, c, false, false);
            c = __builtin_amdgcn_wmma_f32_16x16x32_bf16(false, ahi[kt], false, blo,
                                                        (short)0, c, false, false);
        }
        // ---- store C: lanes 0-15 rows v, lanes 16-31 rows v+8 ----
#pragma unroll
        for (int v = 0; v < 8; ++v) {
            int r = tile_m * 16 + khalf * 8 + v;
            if (r < nrows && ncol < Fout) {
                float val = c[v];
                if (bias) val += bias[ncol];
                Y[(size_t)r * Fout + ncol] = val;
            }
        }
    }
}

// ---- agg = h * dinv^2 (self-loop term; also initializes agg) ---------------
__global__ void gcn_selfterm_kernel(const float* __restrict__ h,
                                    const float* __restrict__ dinv,
                                    float* __restrict__ agg, int total) {
    int i = blockIdx.x * blockDim.x + threadIdx.x;
    if (i >= total) return;
    int node = i >> 7;                    // /128 features
    float di = dinv[node];
    agg[i] = h[i] * di * di;
}

// ---- wave-per-edge scatter: agg[dst] += h[src] * dinv[src]*dinv[dst] -------
__global__ __launch_bounds__(256)
void gcn_scatter_kernel(const float* __restrict__ h,
                        const float* __restrict__ dinv,
                        const int* __restrict__ src,
                        const int* __restrict__ dst,
                        float* __restrict__ agg, int E) {
    int e = blockIdx.x * 8 + (threadIdx.x >> 5);
    if (e >= E) return;
    int lane = threadIdx.x & 31;
    int s = src[e], d = dst[e];
    float nrm = dinv[s] * dinv[d];
    float4 v = ((const float4*)(h + (size_t)s * KDIM))[lane];  // 32 lanes * 4 = 128
    float* a = agg + (size_t)d * KDIM + lane * 4;
    atomicAdd(a + 0, v.x * nrm);
    atomicAdd(a + 1, v.y * nrm);
    atomicAdd(a + 2, v.z * nrm);
    atomicAdd(a + 3, v.w * nrm);
}

// ---- x = relu(x + b[f]) in place -------------------------------------------
__global__ void gcn_bias_relu_kernel(float* __restrict__ x,
                                     const float* __restrict__ b, int total) {
    int i = blockIdx.x * blockDim.x + threadIdx.x;
    if (i >= total) return;
    float v = x[i] + b[i & (KDIM - 1)];
    x[i] = v > 0.0f ? v : 0.0f;
}

// ---------------------------------------------------------------------------
extern "C" void kernel_launch(void* const* d_in, const int* in_sizes, int n_in,
                              void* d_out, int out_size, void* d_ws, size_t ws_size,
                              hipStream_t stream) {
    const float* fts = (const float*)d_in[0];
    const int*   ei  = (const int*)  d_in[1];
    const float* W1  = (const float*)d_in[2];
    const float* b1  = (const float*)d_in[3];
    const float* W2  = (const float*)d_in[4];
    const float* b2  = (const float*)d_in[5];
    const float* Wc  = (const float*)d_in[6];
    const float* bc  = (const float*)d_in[7];

    const int N    = in_sizes[0] / KDIM;
    const int E    = in_sizes[1] / 2;
    const int OUT  = in_sizes[7];                 // 40
    const int OUTp = (OUT + 15) & ~15;            // 48
    const int* src = ei;
    const int* dst = ei + E;

    // ---- workspace layout (floats), everything re-initialized every call ----
    float* ws   = (float*)d_ws;
    float* deg  = ws;                             // N
    float* dinv = ws + N;                         // N
    float* h    = ws + 2 * (size_t)N;             // N*128
    float* agg  = h + (size_t)N * KDIM;           // N*128
    __bf16* wb  = (__bf16*)(agg + (size_t)N * KDIM);
    __bf16* W1hi = wb;
    __bf16* W1lo = W1hi + (size_t)KDIM * KDIM;
    __bf16* W2hi = W1lo + (size_t)KDIM * KDIM;
    __bf16* W2lo = W2hi + (size_t)KDIM * KDIM;
    __bf16* Wchi = W2lo + (size_t)KDIM * KDIM;
    __bf16* Wclo = Wchi + (size_t)OUTp * KDIM;

    float* out = (float*)d_out;                   // [N x OUT]
    float* x2  = out + (size_t)N * OUT;           // [N x 128] (2nd tuple output)

    const int T = 256;
    const int mtiles     = cdiv_i(N, 16);
    const int gemm_grid  = cdiv_i(mtiles, 4);     // 4 waves / block
    const int nf_grid    = cdiv_i(N * KDIM, T);
    const int n_grid     = cdiv_i(N, T);
    const int e_grid     = cdiv_i(E, T);
    const int sc_grid    = cdiv_i(E, 8);          // 8 edges (waves) / block

    // 0) weight prep: transposed bf16 hi/lo splits
    gcn_wsplit_kernel<<<cdiv_i(KDIM * KDIM, T), T, 0, stream>>>(W1, W1hi, W1lo, KDIM, KDIM);
    gcn_wsplit_kernel<<<cdiv_i(KDIM * KDIM, T), T, 0, stream>>>(W2, W2hi, W2lo, KDIM, KDIM);
    gcn_wsplit_kernel<<<cdiv_i(OUTp * KDIM, T), T, 0, stream>>>(Wc, Wchi, Wclo, OUT, OUTp);

    // 1) normalization: deg = 1 + indegree; dinv = deg^-1/2
    gcn_deg_init_kernel<<<n_grid, T, 0, stream>>>(deg, N);
    gcn_deg_accum_kernel<<<e_grid, T, 0, stream>>>(dst, deg, E);
    gcn_dinv_kernel<<<n_grid, T, 0, stream>>>(deg, dinv, N);

    // 2) layer 1: h1 = fts @ W1 ; agg = h1*dinv^2 + scatter ; x1 = relu(agg+b1)
    gcn_gemm_wmma_kernel<<<gemm_grid, 128, 0, stream>>>(fts, W1hi, W1lo, h,
                                                        nullptr, N, KDIM, KDIM / 16);
    gcn_selfterm_kernel<<<nf_grid, T, 0, stream>>>(h, dinv, agg, N * KDIM);
    gcn_scatter_kernel<<<sc_grid, T, 0, stream>>>(h, dinv, src, dst, agg, E);
    gcn_bias_relu_kernel<<<nf_grid, T, 0, stream>>>(agg, b1, N * KDIM);   // agg := x1

    // 3) layer 2: h2 = x1 @ W2 ; accumulate directly into d_out's x2 region
    gcn_gemm_wmma_kernel<<<gemm_grid, 128, 0, stream>>>(agg, W2hi, W2lo, h,
                                                        nullptr, N, KDIM, KDIM / 16);
    gcn_selfterm_kernel<<<nf_grid, T, 0, stream>>>(h, dinv, x2, N * KDIM);
    gcn_scatter_kernel<<<sc_grid, T, 0, stream>>>(h, dinv, src, dst, x2, E);
    gcn_bias_relu_kernel<<<nf_grid, T, 0, stream>>>(x2, b2, N * KDIM);    // x2 final

    // 4) classifier: out = x2 @ Wc + bc   (Fpad=48 tiles, store guarded to 40)
    gcn_gemm_wmma_kernel<<<gemm_grid, 128, 0, stream>>>(x2, Wchi, Wclo, out,
                                                        bc, N, OUT, OUTp / 16);
}
